// MODResBlk_4174708211797
// MI455X (gfx1250) — matrix operand order repaired and checked by reading, not compiled, not run on hardware
//
#include <hip/hip_runtime.h>
#include <hip/hip_bf16.h>

typedef __attribute__((ext_vector_type(16))) __bf16 v16bf;
typedef __attribute__((ext_vector_type(8)))  float  v8f;
typedef unsigned short u16;

#define HH 256
#define WW 256
#define HW 65536
#define CIN 64
#define COUT 128
#define SDIM 64
#define NEG 0.2f
#define INV_SQRT2 0.70710678118654752440f

// ---------- helpers ----------
__device__ __forceinline__ u16 f2bf(float f) {
  unsigned int u = __builtin_bit_cast(unsigned int, f);
  unsigned int lsb = (u >> 16) & 1u;
  u += 0x7fffu + lsb;               // round-to-nearest-even
  return (u16)(u >> 16);
}
__device__ __forceinline__ float lrelu(float x) { return x >= 0.f ? x : NEG * x; }

struct B32x8 { uint4 a, b; };       // 32 bytes == v16bf

__device__ __forceinline__ v16bf frag16(const u16* p) {
  B32x8 t;
  t.a = *(const uint4*)(p);
  t.b = *(const uint4*)(p + 8);
  return __builtin_bit_cast(v16bf, t);
}
__device__ __forceinline__ v8f wmma_bf16(v16bf a, v16bf b, v8f c) {
  return __builtin_amdgcn_wmma_f32_16x16x32_bf16(false, a, false, b, (short)0, c,
                                                 false, false);
}
// A-fragment K mapping within a K=32 block (contiguous: k = 16*(lane/16) + e)
__device__ __forceinline__ int kloc(int lane, int e) {
  return e + ((lane >= 16) ? 16 : 0);
}

// ---------- prep: styles + demod ----------
__global__ __launch_bounds__(256)
void prep_styles(const float* __restrict__ s,
                 const float* __restrict__ a1w, const float* __restrict__ a1b,
                 const float* __restrict__ w1,
                 const float* __restrict__ a2w, const float* __restrict__ a2b,
                 const float* __restrict__ w2,
                 float* __restrict__ style1, float* __restrict__ style2,
                 float* __restrict__ d1, float* __restrict__ d2) {
  __shared__ float st1[8 * CIN];
  __shared__ float st2[8 * COUT];
  int tid = threadIdx.x;
  for (int idx = tid; idx < 8 * CIN; idx += 256) {
    int b = idx / CIN, i = idx % CIN;
    float acc = 0.f;
    for (int k = 0; k < SDIM; ++k) acc += s[b * SDIM + k] * a1w[i * SDIM + k];
    acc += a1b[i] + 1.0f;
    st1[idx] = acc; style1[idx] = acc;
  }
  for (int idx = tid; idx < 8 * COUT; idx += 256) {
    int b = idx / COUT, o = idx % COUT;
    float acc = 0.f;
    for (int k = 0; k < SDIM; ++k) acc += s[b * SDIM + k] * a2w[o * SDIM + k];
    acc += a2b[o] + 1.0f;
    st2[idx] = acc; style2[idx] = acc;
  }
  __syncthreads();
  for (int idx = tid; idx < 8 * COUT; idx += 256) {    // d1
    int b = idx / COUT, o = idx % COUT;
    float acc = 0.f;
    for (int i = 0; i < CIN; ++i) {
      float si = st1[b * CIN + i]; float si2 = si * si;
      for (int t = 0; t < 9; ++t) {
        float wv = w1[(o * CIN + i) * 9 + t];
        acc += wv * wv * si2;
      }
    }
    d1[idx] = rsqrtf(acc + 1e-8f);
  }
  for (int idx = tid; idx < 8 * COUT; idx += 256) {    // d2
    int b = idx / COUT, o = idx % COUT;
    float acc = 0.f;
    for (int i = 0; i < COUT; ++i) {
      float si = st2[b * COUT + i]; float si2 = si * si;
      for (int t = 0; t < 9; ++t) {
        float wv = w2[(o * COUT + i) * 9 + t];
        acc += wv * wv * si2;
      }
    }
    d2[idx] = rsqrtf(acc + 1e-8f);
  }
}

// ---------- weight packers: per-lane WMMA-A fragment order, bf16 ----------
// frag layout: [kstep][mtile(8)][lane(32)][elem(16)], K = kstep*32 + kloc(lane,e)
__global__ __launch_bounds__(256)
void pack_w1(const float* __restrict__ w1, u16* __restrict__ frag) {
  int idx = blockIdx.x * 256 + threadIdx.x;       // 18*8*32*16 = 73728
  int e = idx & 15, lane = (idx >> 4) & 31, mt = (idx >> 9) & 7, ks = idx >> 12;
  int K = ks * 32 + kloc(lane, e);
  int t = K / CIN, i = K % CIN;                   // tap-major K
  int o = mt * 16 + (lane & 15);
  frag[idx] = f2bf(w1[(o * CIN + i) * 9 + t]);
}
__global__ __launch_bounds__(256)
void pack_w2(const float* __restrict__ w2, u16* __restrict__ frag) {
  int idx = blockIdx.x * 256 + threadIdx.x;       // 36*8*32*16 = 147456
  int e = idx & 15, lane = (idx >> 4) & 31, mt = (idx >> 9) & 7, ks = idx >> 12;
  int K = ks * 32 + kloc(lane, e);
  int t = K / COUT, i = K % COUT;
  int o = mt * 16 + (lane & 15);
  frag[idx] = f2bf(w2[(o * COUT + i) * 9 + t]);
}
__global__ __launch_bounds__(256)
void pack_sc(const float* __restrict__ scw, u16* __restrict__ frag) {
  int idx = blockIdx.x * 256 + threadIdx.x;       // 2*8*32*16 = 8192
  int e = idx & 15, lane = (idx >> 4) & 31, mt = (idx >> 9) & 7, ks = idx >> 12;
  int K = ks * 32 + kloc(lane, e);                // channel index
  int o = mt * 16 + (lane & 15);
  frag[idx] = f2bf(scw[o * CIN + K]);
}

// ---------- conv1: 3x3 64->128 modulated; block tile 128 cout x 32 px ----------
__global__ __launch_bounds__(128)
void conv1_kernel(const float* __restrict__ x, const float* __restrict__ noise,
                  const u16* __restrict__ w1frag,
                  const float* __restrict__ style1, const float* __restrict__ style2,
                  const float* __restrict__ d1, const float* __restrict__ nwp,
                  u16* __restrict__ hws) {
  __shared__ __align__(16) u16 act[3 * 34 * CIN];   // halo tile, [row*34+col][ch]
  int bid = blockIdx.x;
  int tw = bid & 7, h0 = (bid >> 3) & 255, b = bid >> 11;
  int w0 = tw * 32;
  int tid = threadIdx.x;

  for (int idx = tid; idx < 3 * 34 * CIN; idx += 128) {
    int ch = idx & 63, colc = (idx >> 6) % 34, row = (idx >> 6) / 34;
    int hh = h0 + row - 1, wc = w0 + colc - 1;
    float v = 0.f;
    if (hh >= 0 && hh < HH && wc >= 0 && wc < WW) {
      float xv = x[((b * CIN + ch) * HH + hh) * WW + wc];
      v = lrelu(xv) * style1[b * CIN + ch];
    }
    act[idx] = f2bf(v);
  }
  __syncthreads();

  int lane = tid & 31, wave = tid >> 5;
  int n = lane & 15, hi = lane >> 4;
  int mtA = wave * 2, mtB = wave * 2 + 1;
  v8f a00 = {}, a01 = {}, a10 = {}, a11 = {};   // [mtile][npart]

  for (int ks = 0; ks < 18; ++ks) {
    int t = ks >> 1, blk = ks & 1;
    int dy = t / 3, dx = t % 3;
    int c0 = blk * 32 + hi * 16;
    v16bf b0 = frag16(&act[(dy * 34 + n + dx) * CIN + c0]);
    v16bf b1 = frag16(&act[(dy * 34 + n + 16 + dx) * CIN + c0]);
    v16bf wa = frag16(w1frag + ((ks * 8 + mtA) * 32 + lane) * 16);
    v16bf wb = frag16(w1frag + ((ks * 8 + mtB) * 32 + lane) * 16);
    a00 = wmma_bf16(wa, b0, a00);
    a01 = wmma_bf16(wa, b1, a01);
    a10 = wmma_bf16(wb, b0, a10);
    a11 = wmma_bf16(wb, b1, a11);
  }

  float nw = nwp[0];
  v8f accs[4] = {a00, a01, a10, a11};
  int mts[4] = {mtA, mtA, mtB, mtB};
  int nps[4] = {0, 16, 0, 16};
  for (int m = 0; m < 4; ++m) {
    int wc = w0 + nps[m] + n;
    float nz = nw * noise[b * HW + h0 * WW + wc];
    int pixbase = ((b * HH + h0) * WW + wc) * COUT;   // NHWC bf16 intermediate
    int cbase = mts[m] * 16 + hi * 8;
    unsigned int pk[4];
    for (int p = 0; p < 4; ++p) {
      int c0i = cbase + 2 * p, c1i = cbase + 2 * p + 1;
      float v0 = accs[m][2 * p]     * d1[b * COUT + c0i] + nz;
      float v1 = accs[m][2 * p + 1] * d1[b * COUT + c1i] + nz;
      v0 = lrelu(v0) * style2[b * COUT + c0i];
      v1 = lrelu(v1) * style2[b * COUT + c1i];
      pk[p] = (unsigned int)f2bf(v0) | ((unsigned int)f2bf(v1) << 16);
    }
    uint4 sv; sv.x = pk[0]; sv.y = pk[1]; sv.z = pk[2]; sv.w = pk[3];
    *(uint4*)(hws + pixbase + cbase) = sv;
  }
}

// ---------- conv2: 3x3 128->128 modulated + fused 1x1 shortcut; 128x32 tile ----------
__global__ __launch_bounds__(128)
void conv2_kernel(const float* __restrict__ x, const float* __restrict__ noise,
                  const u16* __restrict__ hsrc,
                  const u16* __restrict__ w2frag, const u16* __restrict__ scfrag,
                  const float* __restrict__ d2, const float* __restrict__ nwp,
                  float* __restrict__ out) {
  __shared__ __align__(16) u16 hs[3 * 34 * COUT];   // 26112 B
  __shared__ __align__(16) u16 xs[32 * CIN];        // 4096 B
  int bid = blockIdx.x;
  int tw = bid & 7, h0 = (bid >> 3) & 255, b = bid >> 11;
  int w0 = tw * 32;
  int tid = threadIdx.x;

  // stage h tile (NHWC bf16, channel-contiguous b128 loads)
  for (int idx = tid; idx < 102 * 8; idx += 128) {
    int chunk = idx & 7, px = idx >> 3;
    int colc = px % 34, row = px / 34;
    int hh = h0 + row - 1, wc = w0 + colc - 1;
    uint4 v = {0u, 0u, 0u, 0u};
    if (hh >= 0 && hh < HH && wc >= 0 && wc < WW)
      v = *(const uint4*)(hsrc + ((b * HH + hh) * WW + wc) * COUT + chunk * 16);
    *(uint4*)(hs + px * COUT + chunk * 16) = v;
  }
  // stage raw x tile for the 1x1 shortcut (no halo)
  for (int idx = tid; idx < 32 * CIN; idx += 128) {
    int ch = idx & 63, colc = idx >> 6;
    xs[colc * CIN + ch] = f2bf(x[((b * CIN + ch) * HH + h0) * WW + (w0 + colc)]);
  }
  __syncthreads();

  int lane = tid & 31, wave = tid >> 5;
  int n = lane & 15, hi = lane >> 4;
  int mtA = wave * 2, mtB = wave * 2 + 1;
  v8f w00 = {}, w01 = {}, w10 = {}, w11 = {};   // main conv accumulators
  v8f s00 = {}, s01 = {}, s10 = {}, s11 = {};   // shortcut accumulators

  for (int ks = 0; ks < 36; ++ks) {
    int t = ks >> 2, blk = ks & 3;
    int dy = t / 3, dx = t % 3;
    int c0 = blk * 32 + hi * 16;
    v16bf b0 = frag16(&hs[(dy * 34 + n + dx) * COUT + c0]);
    v16bf b1 = frag16(&hs[(dy * 34 + n + 16 + dx) * COUT + c0]);
    v16bf wa = frag16(w2frag + ((ks * 8 + mtA) * 32 + lane) * 16);
    v16bf wb = frag16(w2frag + ((ks * 8 + mtB) * 32 + lane) * 16);
    w00 = wmma_bf16(wa, b0, w00);
    w01 = wmma_bf16(wa, b1, w01);
    w10 = wmma_bf16(wb, b0, w10);
    w11 = wmma_bf16(wb, b1, w11);
  }
  for (int ks = 0; ks < 2; ++ks) {
    int c0 = ks * 32 + hi * 16;
    v16bf b0 = frag16(&xs[n * CIN + c0]);
    v16bf b1 = frag16(&xs[(n + 16) * CIN + c0]);
    v16bf wa = frag16(scfrag + ((ks * 8 + mtA) * 32 + lane) * 16);
    v16bf wb = frag16(scfrag + ((ks * 8 + mtB) * 32 + lane) * 16);
    s00 = wmma_bf16(wa, b0, s00);
    s01 = wmma_bf16(wa, b1, s01);
    s10 = wmma_bf16(wb, b0, s10);
    s11 = wmma_bf16(wb, b1, s11);
  }

  float nw = nwp[0];
  v8f aws[4] = {w00, w01, w10, w11};
  v8f ass[4] = {s00, s01, s10, s11};
  int mts[4] = {mtA, mtA, mtB, mtB};
  int nps[4] = {0, 16, 0, 16};
  for (int m = 0; m < 4; ++m) {
    int wc = w0 + nps[m] + n;
    float nz = nw * noise[b * HW + h0 * WW + wc];
    int cbase = mts[m] * 16 + hi * 8;
    for (int v = 0; v < 8; ++v) {
      int co = cbase + v;
      float val = (aws[m][v] * d2[b * COUT + co] + nz + ass[m][v]) * INV_SQRT2;
      out[(b * COUT + co) * HW + h0 * WW + wc] = val;
    }
  }
}

// ---------- launch ----------
extern "C" void kernel_launch(void* const* d_in, const int* in_sizes, int n_in,
                              void* d_out, int out_size, void* d_ws, size_t ws_size,
                              hipStream_t stream) {
  const float* x    = (const float*)d_in[0];
  const float* s    = (const float*)d_in[1];
  const float* nois = (const float*)d_in[2];
  const float* a1w  = (const float*)d_in[3];
  const float* a1b  = (const float*)d_in[4];
  const float* w1   = (const float*)d_in[5];
  const float* a2w  = (const float*)d_in[6];
  const float* a2b  = (const float*)d_in[7];
  const float* w2   = (const float*)d_in[8];
  const float* nwp  = (const float*)d_in[9];
  const float* scw  = (const float*)d_in[10];
  float* out = (float*)d_out;

  // workspace carve-up (all offsets 16B aligned)
  float* style1 = (float*)d_ws;                  // 512 f32
  float* style2 = style1 + 512;                  // 1024 f32
  float* d1     = style2 + 1024;                 // 1024 f32
  float* d2     = d1 + 1024;                     // 1024 f32
  u16* w1f = (u16*)(d2 + 1024);                  // 73728  u16
  u16* w2f = w1f + 73728;                        // 147456 u16
  u16* scf = w2f + 147456;                       // 8192   u16
  u16* hws = scf + 8192;                         // 8*256*256*128 u16 (NHWC bf16)

  prep_styles<<<1, 256, 0, stream>>>(s, a1w, a1b, w1, a2w, a2b, w2,
                                     style1, style2, d1, d2);
  pack_w1<<<288, 256, 0, stream>>>(w1, w1f);
  pack_w2<<<576, 256, 0, stream>>>(w2, w2f);
  pack_sc<<<32, 256, 0, stream>>>(scw, scf);

  dim3 grid(8 * 256 * 8);   // B * H * (W/32)
  conv1_kernel<<<grid, 128, 0, stream>>>(x, nois, w1f, style1, style2, d1, nwp, hws);
  conv2_kernel<<<grid, 128, 0, stream>>>(x, nois, hws, w2f, scf, d2, nwp, out);
}
